// DeepLeapfrogModel_86088324481809
// MI455X (gfx1250) — matrix operand order repaired and checked by reading, not compiled, not run on hardware
//
#include <hip/hip_runtime.h>
#include <math.h>

// ---------------------------------------------------------------------------
// DeepLeapfrog GNN for MI455X (gfx1250, wave32).
// Dense GEMMs: v_wmma_f32_16x16x32_bf16, f32 accumulate, compile-time K with
// fully unrolled K loop, A tile staged through LDS via async-to-LDS copies
// (double buffered on ASYNCcnt).
// Edge lin_e projections are algebraically refactored to node-level GEMMs.
// ---------------------------------------------------------------------------

#define D128 128
#define HEADS 4

typedef __attribute__((ext_vector_type(16))) __bf16 v16bf;
typedef __attribute__((ext_vector_type(8)))  float  v8f;

#if __has_builtin(__builtin_amdgcn_global_load_async_to_lds_b128)
#define HAVE_ASYNC_LDS 1
#else
#define HAVE_ASYNC_LDS 0
#endif

// exact parameter types per hipcc diagnostic: int4* in AS1 (global), AS3 (LDS)
typedef int v4i_vec __attribute__((vector_size(16)));
typedef v4i_vec __attribute__((address_space(1)))* as1_v4i;
typedef v4i_vec __attribute__((address_space(3)))* as3_v4i;

template <int N>
__device__ __forceinline__ void wait_async() {
#if HAVE_ASYNC_LDS
#if __has_builtin(__builtin_amdgcn_s_wait_asynccnt)
    __builtin_amdgcn_s_wait_asynccnt(N);
#else
    asm volatile("s_wait_asynccnt %0" ::"i"(N) : "memory");
#endif
#endif
}

__device__ __forceinline__ float gelu_f(float x) {
    return 0.5f * x * (1.0f + erff(x * 0.70710678118654752f));
}
__device__ __forceinline__ unsigned f2o(float f) {
    unsigned u = __float_as_uint(f);
    return (u & 0x80000000u) ? ~u : (u | 0x80000000u);
}
__device__ __forceinline__ float o2f(unsigned o) {
    unsigned u = (o & 0x80000000u) ? (o & 0x7fffffffu) : ~o;
    return __uint_as_float(u);
}
__device__ __forceinline__ float4 ld4(const float* __restrict__ base, int n, int lane) {
    return ((const float4*)(base + (size_t)n * D128))[lane];
}

// ---------------------------------------------------------------------------
// WMMA GEMM: C[M x Nc] = act(A[M x K]) @ W[K x Nc] + bias (+res)
// Block = 128 threads (4 waves); block owns a 16-row x 64-col output tile,
// each wave a 16x16 sub-tile. Grid = (M/16, Nc/64). K in {64,128,256}.
// A tile (16x32 f32, 2KB) staged in LDS, shared by all 4 waves; double
// buffered via global_load_async_to_lds_b128 + s_wait_asynccnt.
// Fragment layouts per CDNA5 ISA 7.12.2.
// ---------------------------------------------------------------------------
template <int K, int PRE_GELU, int POST_GELU>
__global__ void gemm_wmma(const float* __restrict__ A, int lda,
                          const float* __restrict__ W, int ldw,
                          const float* __restrict__ bias,
                          const float* __restrict__ residual, int ldr,
                          float* __restrict__ C, int ldc,
                          int M, int Nc) {
    constexpr int NK = K / 32;
    __shared__ float As[2][16 * 32];

    const int tid  = threadIdx.x;
    const int wave = tid >> 5;
    const int lane = tid & 31;
    const int rc   = lane & 15;   // A row / B,D column within tile
    const int half = lane >> 4;
    const int m0 = blockIdx.x * 16;
    const int n0 = blockIdx.y * 64 + wave * 16;

    // A-tile staging: thread t copies 16B: row t/8, floats (t%8)*4 .. +3
    const int lr = tid >> 3;
    const int lc = (tid & 7) * 4;
    const float* Ag = A + (size_t)(m0 + lr) * lda + lc;

    auto issue = [&](int kt) {
        const float* gp = Ag + kt * 32;
        float*       lp = &As[kt & 1][lr * 32 + lc];
#if HAVE_ASYNC_LDS
        __builtin_amdgcn_global_load_async_to_lds_b128(
            (as1_v4i)(uintptr_t)gp, (as3_v4i)(uintptr_t)lp, 0, 0);
#else
        *(float4*)lp = *(const float4*)gp;
#endif
    };

    issue(0);
    v8f acc = {};

#pragma unroll
    for (int kt = 0; kt < NK; ++kt) {
        const int k0 = kt * 32;
        if (kt + 1 < NK) {
            issue(kt + 1);
            __builtin_prefetch(&W[(size_t)(k0 + 32 + half * 16) * ldw + n0], 0, 1);
            wait_async<1>();   // current tile's loads complete (in-order)
        } else {
            wait_async<0>();
        }
        __syncthreads();

        // ---- A fragment from LDS: elems 0..7 -> K=half*8+i ; 8..15 -> K=16+half*8+i
        const float4* A4 = (const float4*)(&As[kt & 1][rc * 32]);
        float4 q0 = A4[half * 2 + 0];
        float4 q1 = A4[half * 2 + 1];
        float4 q2 = A4[half * 2 + 4];
        float4 q3 = A4[half * 2 + 5];
        float fa[16] = {q0.x, q0.y, q0.z, q0.w, q1.x, q1.y, q1.z, q1.w,
                        q2.x, q2.y, q2.z, q2.w, q3.x, q3.y, q3.z, q3.w};
        v16bf a;
#pragma unroll
        for (int i = 0; i < 16; ++i) {
            float v = fa[i];
            if (PRE_GELU) v = gelu_f(v);
            a[i] = (__bf16)v;
        }
        // ---- B fragment from global: elems 0..7 -> K=half*8+i ; 8..15 -> K=16+half*8+i, column = n0+rc
        v16bf b;
#pragma unroll
        for (int i = 0; i < 16; ++i)
            b[i] = (__bf16)W[(size_t)(k0 + half * 8 + (i & 7) + (i >> 3) * 16) * ldw + (n0 + rc)];

        acc = __builtin_amdgcn_wmma_f32_16x16x32_bf16(
            false, a, false, b, (short)0, acc, false, false);
        __syncthreads();
    }

    const int col = n0 + rc;
    const float bv = bias ? bias[col] : 0.0f;
#pragma unroll
    for (int j = 0; j < 8; ++j) {
        int r = m0 + j + half * 8;          // D: VGPR j, lanes16-31 -> M=j+8
        float v = acc[j] + bv;
        if (POST_GELU) v = gelu_f(v);
        if (residual) v += residual[(size_t)r * ldr + col];
        C[(size_t)r * ldc + col] = v;
    }
}

template <int K, int PRE_G, int POST_G>
static inline void launch_gemm(hipStream_t st, const float* A, int lda,
                               const float* W, int ldw, const float* bias,
                               const float* res, int ldr, float* C, int ldc,
                               int M, int Nc) {
    dim3 g((M + 15) / 16, Nc / 64), b(128);
    gemm_wmma<K, PRE_G, POST_G><<<g, b, 0, st>>>(A, lda, W, ldw, bias, res, ldr, C, ldc, M, Nc);
}

// ---------------------------------------------------------------------------
// Time embedding (t_half | t_full) -> temb[2*128]
// ---------------------------------------------------------------------------
__global__ void tembed_kernel(const float* __restrict__ t, const float* __restrict__ freq,
                              const float* __restrict__ W1, const float* __restrict__ b1,
                              const float* __restrict__ W2, const float* __restrict__ b2,
                              float* __restrict__ out) {
    __shared__ float pe[32];
    __shared__ float h1[128];
    float tv = t[0] * (blockIdx.x == 0 ? 0.5f : 1.0f);
    int j = threadIdx.x;
    if (j < 16) {
        float ph = tv * freq[j];
        pe[j]      = sinf(ph);
        pe[j + 16] = cosf(ph);
    }
    __syncthreads();
    float acc = b1[j];
    for (int k = 0; k < 32; ++k) acc += pe[k] * W1[k * 128 + j];
    h1[j] = gelu_f(acc);
    __syncthreads();
    float acc2 = b2[j];
    for (int k = 0; k < 128; ++k) acc2 += h1[k] * W2[k * 128 + j];
    out[blockIdx.x * 128 + j] = acc2;
}

// style = temb @ fcW(128x256) + fcb      (256 threads)
__global__ void style_kernel(const float* __restrict__ temb, const float* __restrict__ fcW,
                             const float* __restrict__ fcb, float* __restrict__ style) {
    int j = threadIdx.x;
    float acc = fcb[j];
    for (int k = 0; k < 128; ++k) acc += temb[k] * fcW[k * 256 + j];
    style[j] = acc;
}

// AdaGN: 8 groups of 16 channels. One node per 128-thread block.
__global__ void adagn_kernel(const float* __restrict__ x, const float* __restrict__ gw,
                             const float* __restrict__ gb, const float* __restrict__ style,
                             float* __restrict__ out) {
    int node = blockIdx.x, c = threadIdx.x;
    float v = x[(size_t)node * D128 + c];
    float s = v;
    for (int off = 1; off < 16; off <<= 1) s += __shfl_xor(s, off, 16);
    float mu = s * (1.0f / 16.0f);
    float d = v - mu;
    float vs = d * d;
    for (int off = 1; off < 16; off <<= 1) vs += __shfl_xor(vs, off, 16);
    float nrm = d * rsqrtf(vs * (1.0f / 16.0f) + 1e-5f);
    float g = style[c], b = style[128 + c];
    out[(size_t)node * D128 + c] = (nrm * gw[c] + gb[c]) * (1.0f + g) + b;
}

// LayerNorm over 128 channels; one node per wave, 8 waves per block.
__global__ void ln_kernel(const float* __restrict__ x, const float* __restrict__ w,
                          const float* __restrict__ b, float* __restrict__ out, int n) {
    int wid = blockIdx.x * 8 + (threadIdx.x >> 5);
    if (wid >= n) return;
    int lane = threadIdx.x & 31;
    float4 v = ld4(x, wid, lane);
    float s = v.x + v.y + v.z + v.w;
    for (int off = 1; off < 32; off <<= 1) s += __shfl_xor(s, off, 32);
    float mu = s * (1.0f / 128.0f);
    float d0 = v.x - mu, d1 = v.y - mu, d2 = v.z - mu, d3 = v.w - mu;
    float vs = d0 * d0 + d1 * d1 + d2 * d2 + d3 * d3;
    for (int off = 1; off < 32; off <<= 1) vs += __shfl_xor(vs, off, 32);
    float inv = rsqrtf(vs * (1.0f / 128.0f) + 1e-5f);
    float4 w4 = ((const float4*)w)[lane];
    float4 b4 = ((const float4*)b)[lane];
    float4 o;
    o.x = d0 * inv * w4.x + b4.x;
    o.y = d1 * inv * w4.y + b4.y;
    o.z = d2 * inv * w4.z + b4.z;
    o.w = d3 * inv * w4.w + b4.w;
    ((float4*)(out + (size_t)wid * D128))[lane] = o;
}

// Node encoders, first layers (tiny K): pre = in @ W + b
__global__ void encode_pre(const float* __restrict__ x, const float* __restrict__ WE,
                           const float* __restrict__ bE, const float* __restrict__ WH,
                           const float* __restrict__ bH, float* __restrict__ preE,
                           float* __restrict__ preH, int n) {
    int id = blockIdx.x * blockDim.x + threadIdx.x;
    if (id >= n * D128) return;
    int node = id >> 7, j = id & 127;
    float Ez = x[node * 4 + 0], Hx = x[node * 4 + 1], Hy = x[node * 4 + 2], ep = x[node * 4 + 3];
    preE[id] = Ez * WE[j] + ep * WE[128 + j] + bE[j];
    preH[id] = Hx * WH[j] + Hy * WH[128 + j] + ep * WH[256 + j] + bH[j];
}

// ---------------------------------------------------------------------------
// Per-edge e_all (projected edge features), refactored to gathers of node GEMMs
// MODE 0: geo H   1: geo E   2: gnn H   3: gnn E
// ---------------------------------------------------------------------------
template <int MODE>
__device__ __forceinline__ float4 compute_ea(int s, int d, int lane,
                                             const float* __restrict__ pos,
                                             const float* __restrict__ xin,
                                             const float* __restrict__ P1,
                                             const float* __restrict__ P2,
                                             const float* __restrict__ Wex) {
    int c = lane * 4;
    float dx = pos[2 * s] - pos[2 * d];
    float dy = pos[2 * s + 1] - pos[2 * d + 1];
    float r2 = dx * dx + dy * dy;
    float ds = (MODE < 2) ? fmaxf(r2, 1e-8f) : (r2 + 1e-8f);
    float4 p1s = ld4(P1, s, lane), p1d = ld4(P1, d, lane);
    float4 ea;
    if (MODE == 0) {
        float4 p2s = ld4(P2, s, lane), p2d = ld4(P2, d, lane);
        float c1 = -dy / ds, c2 = dx / ds;
        ea.x = c1 * (p1s.x - p1d.x) + c2 * (p2s.x - p2d.x);
        ea.y = c1 * (p1s.y - p1d.y) + c2 * (p2s.y - p2d.y);
        ea.z = c1 * (p1s.z - p1d.z) + c2 * (p2s.z - p2d.z);
        ea.w = c1 * (p1s.w - p1d.w) + c2 * (p2s.w - p2d.w);
    } else if (MODE == 1) {
        float4 p2s = ld4(P2, s, lane), p2d = ld4(P2, d, lane);
        float c1 = dx / ds, c2 = dy / ds;
        float epe = 0.5f * (xin[4 * s + 3] + xin[4 * d + 3]);
        ea.x = c1 * (p2s.x + p2d.x) - c2 * (p1s.x + p1d.x) + epe * Wex[c + 0];
        ea.y = c1 * (p2s.y + p2d.y) - c2 * (p1s.y + p1d.y) + epe * Wex[c + 1];
        ea.z = c1 * (p2s.z + p2d.z) - c2 * (p1s.z + p1d.z) + epe * Wex[c + 2];
        ea.w = c1 * (p2s.w + p2d.w) - c2 * (p1s.w + p1d.w) + epe * Wex[c + 3];
    } else if (MODE == 2) {
        float gx = dx / ds, gy = dy / ds;
        ea.x = (p1s.x - p1d.x) - gy * Wex[c + 0] + gx * Wex[128 + c + 0];
        ea.y = (p1s.y - p1d.y) - gy * Wex[c + 1] + gx * Wex[128 + c + 1];
        ea.z = (p1s.z - p1d.z) - gy * Wex[c + 2] + gx * Wex[128 + c + 2];
        ea.w = (p1s.w - p1d.w) - gy * Wex[c + 3] + gx * Wex[128 + c + 3];
    } else {
        float gx = dx / ds, gy = dy / ds;
        float epe = 0.5f * (xin[4 * s + 3] + xin[4 * d + 3]);
        ea.x = (p1s.x + p1d.x) + gx * Wex[c + 0] + gy * Wex[128 + c + 0] + epe * Wex[256 + c + 0];
        ea.y = (p1s.y + p1d.y) + gx * Wex[c + 1] + gy * Wex[128 + c + 1] + epe * Wex[256 + c + 1];
        ea.z = (p1s.z + p1d.z) + gx * Wex[c + 2] + gy * Wex[128 + c + 2] + epe * Wex[256 + c + 2];
        ea.w = (p1s.w + p1d.w) + gx * Wex[c + 3] + gy * Wex[128 + c + 3] + epe * Wex[256 + c + 3];
    }
    return ea;
}

// e_loop = mean_e(e_all). LDS-staged partial sums (ds_add_f32) + per-block flush.
template <int MODE>
__global__ void eloop_accum(const int* __restrict__ src, const int* __restrict__ dst, int E,
                            const float* __restrict__ pos, const float* __restrict__ xin,
                            const float* __restrict__ P1, const float* __restrict__ P2,
                            const float* __restrict__ Wex, float* __restrict__ gsum) {
    __shared__ float ssum[128];
    if (threadIdx.x < 128) ssum[threadIdx.x] = 0.0f;
    __syncthreads();
    int wid = (blockIdx.x * blockDim.x + threadIdx.x) >> 5;
    int lane = threadIdx.x & 31;
    if (wid < E) {
        float4 ea = compute_ea<MODE>(src[wid], dst[wid], lane, pos, xin, P1, P2, Wex);
        int c = lane * 4;
        atomicAdd(&ssum[c + 0], ea.x);
        atomicAdd(&ssum[c + 1], ea.y);
        atomicAdd(&ssum[c + 2], ea.z);
        atomicAdd(&ssum[c + 3], ea.w);
    }
    __syncthreads();
    if (threadIdx.x < 128) atomicAdd(&gsum[threadIdx.x], ssum[threadIdx.x]);
}

__global__ void eloop_fin(const float* __restrict__ gsum, float* __restrict__ eloop, int E) {
    eloop[threadIdx.x] = gsum[threadIdx.x] / (float)E;
}

// aggr <- broadcast gat bias ; denom/mmax <- 0
__global__ void init_aggr(const float* __restrict__ bias, float* __restrict__ aggr,
                          float* __restrict__ denom, unsigned* __restrict__ mmax, int n) {
    int id = blockIdx.x * blockDim.x + threadIdx.x;
    if (id >= n * D128) return;
    int node = id >> 7, c = id & 127;
    aggr[id] = bias[c];
    if (c < 4) {
        denom[node * 4 + c] = 0.0f;
        mmax[node * 4 + c]  = 0u;
    }
}

// Pass A: z = leaky(xl[s]+xr[d]+e_all), logits per head, segment max (ordered atomics)
template <int MODE>
__global__ void edge_logits(const int* __restrict__ src, const int* __restrict__ dst,
                            int E, int Etot, const float* __restrict__ pos,
                            const float* __restrict__ xin, const float* __restrict__ xl,
                            const float* __restrict__ xr, const float* __restrict__ P1,
                            const float* __restrict__ P2, const float* __restrict__ Wex,
                            const float* __restrict__ eloop, const float* __restrict__ att,
                            float* __restrict__ logits, unsigned* __restrict__ mmax) {
    int wid = (blockIdx.x * blockDim.x + threadIdx.x) >> 5;
    if (wid >= Etot) return;
    int lane = threadIdx.x & 31;
    bool self = wid >= E;
    int s, d;
    if (self) { s = d = wid - E; } else { s = src[wid]; d = dst[wid]; }

    float4 ea = self ? ((const float4*)eloop)[lane]
                     : compute_ea<MODE>(s, d, lane, pos, xin, P1, P2, Wex);
    float4 vl = ld4(xl, s, lane);
    float4 vr = ld4(xr, d, lane);
    float4 at = ((const float4*)att)[lane];

    float z0 = vl.x + vr.x + ea.x; z0 = z0 > 0.0f ? z0 : 0.2f * z0;
    float z1 = vl.y + vr.y + ea.y; z1 = z1 > 0.0f ? z1 : 0.2f * z1;
    float z2 = vl.z + vr.z + ea.z; z2 = z2 > 0.0f ? z2 : 0.2f * z2;
    float z3 = vl.w + vr.w + ea.w; z3 = z3 > 0.0f ? z3 : 0.2f * z3;
    float p = z0 * at.x + z1 * at.y + z2 * at.z + z3 * at.w;
    p += __shfl_xor(p, 1, 8);
    p += __shfl_xor(p, 2, 8);
    p += __shfl_xor(p, 4, 8);
    if ((lane & 7) == 0) {
        int h = lane >> 3;
        logits[(size_t)wid * 4 + h] = p;
        atomicMax(&mmax[(size_t)d * 4 + h], f2o(p));
    }
}

// Pass B: a = exp(logit - m[d]); denom[d] += a
__global__ void edge_exp(const float* __restrict__ logits, const int* __restrict__ dst,
                         int E, int Etot, const unsigned* __restrict__ mmax,
                         float* __restrict__ aexp, float* __restrict__ denom) {
    int idx = blockIdx.x * blockDim.x + threadIdx.x;
    if (idx >= Etot * 4) return;
    int e = idx >> 2, h = idx & 3;
    int d = (e < E) ? dst[e] : (e - E);
    float m = o2f(mmax[d * 4 + h]);
    float a = __expf(logits[idx] - m);
    aexp[idx] = a;
    atomicAdd(&denom[d * 4 + h], a);
}

// Pass C: aggr[d] += xl[s] * alpha   (float atomics, 4 ch/lane)
__global__ void edge_aggr(const int* __restrict__ src, const int* __restrict__ dst,
                          int E, int Etot, const float* __restrict__ aexp,
                          const float* __restrict__ denom, const float* __restrict__ xl,
                          float* __restrict__ aggr) {
    int wid = (blockIdx.x * blockDim.x + threadIdx.x) >> 5;
    if (wid >= Etot) return;
    int lane = threadIdx.x & 31;
    bool self = wid >= E;
    int s, d;
    if (self) { s = d = wid - E; } else { s = src[wid]; d = dst[wid]; }
    int h = lane >> 3;
    float alpha = aexp[(size_t)wid * 4 + h] / (denom[(size_t)d * 4 + h] + 1e-16f);
    float4 v = ld4(xl, s, lane);
    float* o = aggr + (size_t)d * D128 + lane * 4;
    atomicAdd(o + 0, v.x * alpha);
    atomicAdd(o + 1, v.y * alpha);
    atomicAdd(o + 2, v.z * alpha);
    atomicAdd(o + 3, v.w * alpha);
}

// Decoder heads: Ez (64->1), HxHy (64->2) from gelu'd hidden states
__global__ void decode_out(const float* __restrict__ g1E, const float* __restrict__ g1H,
                           const float* __restrict__ WE2, const float* __restrict__ bE2,
                           const float* __restrict__ WH2, const float* __restrict__ bH2,
                           float* __restrict__ out, int n) {
    int i = blockIdx.x * blockDim.x + threadIdx.x;
    if (i >= n) return;
    const float* ge = g1E + (size_t)i * 64;
    const float* gh = g1H + (size_t)i * 64;
    float aE = bE2[0], aHx = bH2[0], aHy = bH2[1];
    for (int k = 0; k < 64; ++k) {
        aE  += ge[k] * WE2[k];
        aHx += gh[k] * WH2[k * 2 + 0];
        aHy += gh[k] * WH2[k * 2 + 1];
    }
    out[(size_t)i * 3 + 0] = aE;
    out[(size_t)i * 3 + 1] = aHx;
    out[(size_t)i * 3 + 2] = aHy;
}

// ---------------------------------------------------------------------------
// Host driver
// ---------------------------------------------------------------------------
extern "C" void kernel_launch(void* const* d_in, const int* in_sizes, int n_in,
                              void* d_out, int out_size, void* d_ws, size_t ws_size,
                              hipStream_t stream) {
    const int N = in_sizes[0] / 4;
    const int E = in_sizes[n_in - 1] / 2;
    const int Etot = E + N;

    auto F = [&](int i) { return (const float*)d_in[i]; };
    const int* ei  = (const int*)d_in[n_in - 1];
    const int* src = ei;
    const int* dst = ei + E;
    const float* xin = F(0);
    const float* pos = F(1);
    const float* tin = F(2);

    // ---- workspace carve (f32 units, 1KB aligned) ----
    size_t o = 0;
    auto alloc = [&](size_t nelem) -> float* {
        float* p = (float*)d_ws + o;
        o += (nelem + 255) & ~(size_t)255;
        return p;
    };
    float* temb  = alloc(256);            // [t_half | t_full]
    float* style = alloc(256);
    float* esum  = alloc(128);
    float* eloop = alloc(128);
    float* hE    = alloc((size_t)N * 128);
    float* hH    = alloc((size_t)N * 128);
    float* hs    = alloc((size_t)N * 128);
    float* xl    = alloc((size_t)N * 128);
    float* xr    = alloc((size_t)N * 128);
    float* P1    = alloc((size_t)N * 128);
    float* P2    = alloc((size_t)N * 128);
    float* aggr  = alloc((size_t)N * 128);
    float* hln   = alloc((size_t)N * 128);
    float* mid   = alloc((size_t)N * 256);
    float* denom = alloc((size_t)N * 4);
    unsigned* mmax = (unsigned*)alloc((size_t)N * 4);
    float* logits = alloc((size_t)Etot * 4);
    float* aexp   = alloc((size_t)Etot * 4);
    float* d1E    = alloc((size_t)N * 64);
    float* d1H    = alloc((size_t)N * 64);
    (void)ws_size; (void)out_size;

    const int NB128 = (N * 128 + 255) / 256;
    const int EB    = (int)(((size_t)E * 32 + 255) / 256);
    const int TB    = (int)(((size_t)Etot * 32 + 255) / 256);

    // ---- time embeddings (t_half, t_full) ----
    tembed_kernel<<<2, 128, 0, stream>>>(tin, F(3), F(12), F(13), F(14), F(15), temb);

    // ---- node encoders ----
    encode_pre<<<NB128, 256, 0, stream>>>(xin, F(4), F(5), F(8), F(9), hs, xl, N);
    launch_gemm<128, 1, 0>(stream, hs, 128, F(6), 128, F(7), nullptr, 0, hE, 128, N, 128);
    launch_gemm<128, 1, 0>(stream, xl, 128, F(10), 128, F(11), nullptr, 0, hH, 128, N, 128);

    // ---- one GNN/GEO block ----
    auto run_block = [&](int base, int mode, int mid_dim, float* target,
                         const float* source, const float* tv) {
        const float *gnw = F(base + 0), *gnb = F(base + 1), *fcW = F(base + 2), *fcb = F(base + 3);
        const float *lW = F(base + 4), *lb = F(base + 5), *rW = F(base + 6), *rb = F(base + 7);
        const float *eW = F(base + 8), *att = F(base + 9), *gbias = F(base + 10);
        const float *poW = F(base + 11), *pob = F(base + 12);
        const float *lnw = F(base + 13), *lnb = F(base + 14);
        const float *m1W = F(base + 15), *m1b = F(base + 16);
        const float *m2W = F(base + 17), *m2b = F(base + 18);

        style_kernel<<<1, 256, 0, stream>>>(tv, fcW, fcb, style);
        adagn_kernel<<<N, 128, 0, stream>>>(source, gnw, gnb, style, hs);

        launch_gemm<128, 0, 0>(stream, hs, 128, lW, 128, lb, nullptr, 0, xl, 128, N, 128);
        launch_gemm<128, 0, 0>(stream, hs, 128, rW, 128, rb, nullptr, 0, xr, 128, N, 128);

        const float* Wex = nullptr;
        if (mode == 0) {                         // ea = [-dy*gm | dx*gm], gm=(hs_s-hs_d)/ds
            launch_gemm<128, 0, 0>(stream, hs, 128, eW, 128, nullptr, nullptr, 0, P1, 128, N, 128);
            launch_gemm<128, 0, 0>(stream, hs, 128, eW + 128 * 128, 128, nullptr, nullptr, 0, P2, 128, N, 128);
        } else if (mode == 1) {                  // QX = hs[:, :64]@We[:64], QY = hs[:,64:]@We[:64]
            launch_gemm<64, 0, 0>(stream, hs, 128, eW, 128, nullptr, nullptr, 0, P1, 128, N, 128);
            launch_gemm<64, 0, 0>(stream, hs + 64, 128, eW, 128, nullptr, nullptr, 0, P2, 128, N, 128);
            Wex = eW + 64 * 128;                 // eps row
        } else {                                 // phys projection + geometric rows
            launch_gemm<128, 0, 0>(stream, hs, 128, eW, 128, nullptr, nullptr, 0, P1, 128, N, 128);
            Wex = eW + 128 * 128;                // ortho/geom/eps rows
        }

        (void)hipMemsetAsync(esum, 0, 128 * sizeof(float), stream);
        switch (mode) {
            case 0: eloop_accum<0><<<EB, 256, 0, stream>>>(src, dst, E, pos, xin, P1, P2, Wex, esum); break;
            case 1: eloop_accum<1><<<EB, 256, 0, stream>>>(src, dst, E, pos, xin, P1, P2, Wex, esum); break;
            case 2: eloop_accum<2><<<EB, 256, 0, stream>>>(src, dst, E, pos, xin, P1, P2, Wex, esum); break;
            default: eloop_accum<3><<<EB, 256, 0, stream>>>(src, dst, E, pos, xin, P1, P2, Wex, esum); break;
        }
        eloop_fin<<<1, 128, 0, stream>>>(esum, eloop, E);

        init_aggr<<<NB128, 256, 0, stream>>>(gbias, aggr, denom, mmax, N);
        switch (mode) {
            case 0: edge_logits<0><<<TB, 256, 0, stream>>>(src, dst, E, Etot, pos, xin, xl, xr, P1, P2, Wex, eloop, att, logits, mmax); break;
            case 1: edge_logits<1><<<TB, 256, 0, stream>>>(src, dst, E, Etot, pos, xin, xl, xr, P1, P2, Wex, eloop, att, logits, mmax); break;
            case 2: edge_logits<2><<<TB, 256, 0, stream>>>(src, dst, E, Etot, pos, xin, xl, xr, P1, P2, Wex, eloop, att, logits, mmax); break;
            default: edge_logits<3><<<TB, 256, 0, stream>>>(src, dst, E, Etot, pos, xin, xl, xr, P1, P2, Wex, eloop, att, logits, mmax); break;
        }
        edge_exp<<<(Etot * 4 + 255) / 256, 256, 0, stream>>>(logits, dst, E, Etot, mmax, aexp, denom);
        edge_aggr<<<TB, 256, 0, stream>>>(src, dst, E, Etot, aexp, denom, xl, aggr);

        // post: (aggr@W+b) -> LN -> MLP -> residual add into target (in place)
        launch_gemm<128, 0, 0>(stream, aggr, 128, poW, 128, pob, nullptr, 0, P1, 128, N, 128);
        ln_kernel<<<(N + 7) / 8, 256, 0, stream>>>(P1, lnw, lnb, hln, N);
        launch_gemm<128, 0, 0>(stream, hln, 128, m1W, mid_dim, m1b, nullptr, 0, mid, mid_dim, N, mid_dim);
        if (mid_dim == 128)
            launch_gemm<128, 1, 0>(stream, mid, 128, m2W, 128, m2b, target, 128, target, 128, N, 128);
        else
            launch_gemm<256, 1, 0>(stream, mid, 256, m2W, 128, m2b, target, 128, target, 128, N, 128);
    };

    run_block(16, 0, 128, hH, hE, temb);         // geo1_H  (t_half)
    run_block(35, 1, 128, hE, hH, temb);         // geo2_E  (t_half)
    run_block(54, 2, 256, hH, hE, temb + 128);   // blk3_H  (t_full)
    run_block(73, 3, 256, hE, hH, temb + 128);   // blk4_E  (t_full)

    // ---- decoders ----
    ln_kernel<<<(N + 7) / 8, 256, 0, stream>>>(hE, F(92), F(93), hln, N);
    launch_gemm<128, 0, 1>(stream, hln, 128, F(94), 64, F(95), nullptr, 0, d1E, 64, N, 64);
    ln_kernel<<<(N + 7) / 8, 256, 0, stream>>>(hH, F(98), F(99), hln, N);
    launch_gemm<128, 0, 1>(stream, hln, 128, F(100), 64, F(101), nullptr, 0, d1H, 64, N, 64);
    decode_out<<<(N + 255) / 256, 256, 0, stream>>>(d1E, d1H, F(96), F(97), F(102), F(103),
                                                    (float*)d_out, N);
}
